// MuiltiHeadAttention_59502476918997
// MI455X (gfx1250) — matrix operand order repaired and checked by reading, not compiled, not run on hardware
//
#include <hip/hip_runtime.h>

typedef __attribute__((ext_vector_type(16))) __bf16 v16bf;
typedef __attribute__((ext_vector_type(8)))  float  v8f;

union FragU { unsigned int u[8]; v16bf v; };

__device__ __forceinline__ unsigned short f32_to_bf16(float f) {
  unsigned int u = __float_as_uint(f);
  u += 0x7FFFu + ((u >> 16) & 1u);   // round-to-nearest-even
  return (unsigned short)(u >> 16);
}

// CDNA5 async global->LDS copy, 16B per lane, tracked by ASYNCcnt (in-order completion).
__device__ __forceinline__ void async_copy_b128(unsigned lds_off, const void* gaddr) {
  asm volatile("global_load_async_to_lds_b128 %0, %1, off"
               :: "v"(lds_off), "v"((unsigned long long)gaddr) : "memory");
}
template <int N>
__device__ __forceinline__ void wait_async_le() {
  asm volatile("s_wait_asynccnt %0" :: "n"(N) : "memory");
}

// ---------------------------------------------------------------- converts
__global__ __launch_bounds__(256)
void cvt_f32_to_bf16(const float* __restrict__ src, unsigned short* __restrict__ dst, int n) {
  for (int i = blockIdx.x * blockDim.x + threadIdx.x; i < n; i += gridDim.x * blockDim.x)
    dst[i] = f32_to_bf16(src[i]);
}

// dst[N][K] = bf16(src[K][N])  (transpose weights once so GEMM B-tiles are K-contiguous)
__global__ __launch_bounds__(256)
void cvt_f32_to_bf16_T(const float* __restrict__ src, unsigned short* __restrict__ dst,
                       int R, int C) {
  for (int i = blockIdx.x * blockDim.x + threadIdx.x; i < R * C; i += gridDim.x * blockDim.x) {
    int r = i / C, c = i % C;
    dst[(size_t)c * R + r] = f32_to_bf16(src[i]);
  }
}

// ---------------------------------------------------------------- GEMM
// C[M,N] = A[M,K] * Bt[N,K]^T (+bias), bf16 in, f32 accum, double-buffered async LDS pipeline.
// mode 0: f32 row-major out.  mode 1: bf16 out -> [B,H,S,D].  mode 2: bf16 out -> [B,H,D,S].
#define BM 128
#define BN 128
#define BK 64

__device__ __forceinline__ void gemm_stage(const unsigned short* __restrict__ A,
                                           const unsigned short* __restrict__ Bt,
                                           unsigned short* As, unsigned short* Bs,
                                           int m0, int n0, int Kd, int k0, int tid) {
#pragma unroll
  for (int i = 0; i < 4; ++i) {                 // 256 thr * 4 * 8 elems = 128x64 tile each
    int e = (tid + i * 256) * 8;
    int r = e / BK, c = e % BK;
    async_copy_b128((unsigned)(size_t)&As[e], &A [(size_t)(m0 + r) * Kd + k0 + c]);
    async_copy_b128((unsigned)(size_t)&Bs[e], &Bt[(size_t)(n0 + r) * Kd + k0 + c]);
  }
}

__global__ __launch_bounds__(256)
void gemm_bf16(const unsigned short* __restrict__ A,
               const unsigned short* __restrict__ Bt,   // [N][K]
               const float* __restrict__ bias,
               float* __restrict__ outF,
               unsigned short* __restrict__ outQ,
               int M, int N, int Kd, int mode) {
  __shared__ __align__(16) unsigned short As[2][BM * BK];   // 2 x 16 KB
  __shared__ __align__(16) unsigned short Bs[2][BN * BK];   // 2 x 16 KB
  const int tid  = threadIdx.x;
  const int lane = tid & 31;
  const int w    = tid >> 5;
  const int l16  = lane & 15, half = lane >> 4;
  const int wm   = w & 1, wn = w >> 1;          // 2x4 wave grid -> 64x32 per wave
  const int m0   = blockIdx.x * BM, n0 = blockIdx.y * BN;

  v8f acc[4][2] = {};

  const int T = Kd / BK;
  gemm_stage(A, Bt, As[0], Bs[0], m0, n0, Kd, 0, tid);      // prologue: tile 0 in flight

  for (int kt = 0; kt < T; ++kt) {
    const int cur = kt & 1;
    if (kt + 1 < T) {                           // overlap next tile's DMA with this compute
      gemm_stage(A, Bt, As[cur ^ 1], Bs[cur ^ 1], m0, n0, Kd, (kt + 1) * BK, tid);
      wait_async_le<8>();                       // tile kt done; tile kt+1 (8 ops) in flight
    } else {
      wait_async_le<0>();
    }
    __syncthreads();                            // tile kt resident for all waves

    const unsigned short* Ac = As[cur];
    const unsigned short* Bc = Bs[cur];
#pragma unroll
    for (int c = 0; c < 2; ++c) {               // two K=32 chunks per staged tile
      v16bf af[4];
#pragma unroll
      for (int mt = 0; mt < 4; ++mt) {          // A frag: row = lane%16, paired-K layout
        int row = wm * 64 + mt * 16 + l16;
        FragU u;
#pragma unroll
        for (int j = 0; j < 8; ++j) {
          int kk = half * 16 + 2 * j;
          u.u[j] = *(const unsigned int*)&Ac[row * BK + c * 32 + kk];
        }
        af[mt] = u.v;
      }
      v16bf bfr[2];
#pragma unroll
      for (int nt = 0; nt < 2; ++nt) {          // B frag: col = lane%16, K = half*16 + i
        int col = wn * 32 + nt * 16 + l16;
        FragU u;
#pragma unroll
        for (int j = 0; j < 8; ++j)
          u.u[j] = *(const unsigned int*)&Bc[col * BK + c * 32 + half * 16 + 2 * j];
        bfr[nt] = u.v;
      }
#pragma unroll
      for (int mt = 0; mt < 4; ++mt)
#pragma unroll
        for (int nt = 0; nt < 2; ++nt)
          acc[mt][nt] = __builtin_amdgcn_wmma_f32_16x16x32_bf16(
              false, af[mt], false, bfr[nt], (short)0, acc[mt][nt], false, false);
    }
    __syncthreads();                            // all waves done reading tile kt
  }

#pragma unroll
  for (int mt = 0; mt < 4; ++mt) {
#pragma unroll
    for (int nt = 0; nt < 2; ++nt) {
      int col  = n0 + wn * 32 + nt * 16 + l16;
      float bv = bias[col];
#pragma unroll
      for (int r = 0; r < 8; ++r) {             // C layout: row = base + half*8 + r
        int row   = m0 + wm * 64 + mt * 16 + half * 8 + r;
        float val = acc[mt][nt][r] + bv;
        int b = row >> 11, s = row & 2047;
        int h = col >> 6,  d = col & 63;
        if (mode == 0) {
          outF[(size_t)row * N + col] = val;
        } else if (mode == 1) {                 // [B,H,S,D]
          outQ[(((size_t)(b * 16 + h) * 2048) + s) * 64 + d] = f32_to_bf16(val);
        } else {                                // [B,H,D,S]  (V stored d-major)
          outQ[(((size_t)(b * 16 + h) * 64) + d) * 2048 + s] = f32_to_bf16(val);
        }
      }
    }
  }
}

// ---------------------------------------------------------------- attention
#define S_LEN 2048
#define HD 64

__global__ __launch_bounds__(256)
void attn_kernel(const unsigned short* __restrict__ Q,   // [BH][S][D]
                 const unsigned short* __restrict__ K,   // [BH][S][D]
                 const unsigned short* __restrict__ Vt,  // [BH][D][S]
                 unsigned short* __restrict__ O) {       // [B][S][E]
  __shared__ __align__(16) unsigned short Ks[2][32 * HD];     // key tiles [key][d], 2 x 4 KB
  __shared__ __align__(16) unsigned short Vs[2][HD * 32];     // value tiles [d][key], 2 x 4 KB
  __shared__ __align__(16) unsigned short Pl[8 * 16 * 32];    // per-wave P tiles, 8 KB
  const int tid  = threadIdx.x;
  const int lane = tid & 31, w = tid >> 5;
  const int l16  = lane & 15, half = lane >> 4;
  const int bh   = blockIdx.x;
  const int q0   = blockIdx.y * 128 + w * 16;   // 16 query rows per wave
  const size_t baseK = (size_t)bh * S_LEN * HD;
  const size_t baseV = (size_t)bh * HD * S_LEN;
  const int pbase = w * 16 * 32;

  // Q as two 16x32 A fragments (global, contiguous 32B/lane)
  v16bf qa[2];
#pragma unroll
  for (int c = 0; c < 2; ++c) {
    FragU u;
#pragma unroll
    for (int j = 0; j < 8; ++j) {
      int kk = half * 16 + 2 * j;
      u.u[j] = *(const unsigned int*)&Q[baseK + (size_t)(q0 + l16) * HD + c * 32 + kk];
    }
    qa[c] = u.v;
  }

  v8f oacc[4] = {};
  float mrow[8], lrow[8];
#pragma unroll
  for (int r = 0; r < 8; ++r) { mrow[r] = -3.0e38f; lrow[r] = 0.0f; }

  const int kkey = tid >> 3, kch = tid & 7;     // K staging: 32 keys x 8 chunks of 8
  const int vd   = tid >> 2, vch = tid & 3;     // V staging: 64 d   x 4 chunks of 8

  auto stage_kv = [&](int kt, int buf) {
    async_copy_b128((unsigned)(size_t)&Ks[buf][kkey * HD + kch * 8],
                    &K[baseK + (size_t)(kt * 32 + kkey) * HD + kch * 8]);
    async_copy_b128((unsigned)(size_t)&Vs[buf][vd * 32 + vch * 8],
                    &Vt[baseV + (size_t)vd * S_LEN + kt * 32 + vch * 8]);
  };

  const int T = S_LEN / 32;
  stage_kv(0, 0);                               // prologue

  for (int kt = 0; kt < T; ++kt) {
    const int cur = kt & 1;
    if (kt + 1 < T) {                           // pipeline next K/V tile
      stage_kv(kt + 1, cur ^ 1);
      wait_async_le<2>();                       // tile kt done; kt+1 (2 ops) in flight
    } else {
      wait_async_le<0>();
    }
    __syncthreads();

    // scores: S = Q * K^T  (B frag from LDS: col = key, contraction = d, contiguous)
    v8f sc[2];
#pragma unroll
    for (int ns = 0; ns < 2; ++ns) {
      int krow = ns * 16 + l16;
      v8f z = {};
#pragma unroll
      for (int c = 0; c < 2; ++c) {
        FragU u;
#pragma unroll
        for (int j = 0; j < 8; ++j)
          u.u[j] = *(const unsigned int*)&Ks[cur][krow * HD + c * 32 + half * 16 + 2 * j];
        z = __builtin_amdgcn_wmma_f32_16x16x32_bf16(false, qa[c], false, u.v,
                                                    (short)0, z, false, false);
      }
      sc[ns] = z;
    }

    // streaming softmax (row = query, spread over 16-lane half-waves)
    float alpha[8];
#pragma unroll
    for (int r = 0; r < 8; ++r) {
      sc[0][r] *= 0.125f; sc[1][r] *= 0.125f;   // 1/sqrt(64)
      float t = fmaxf(sc[0][r], sc[1][r]);
      t = fmaxf(t, __shfl_xor(t, 1, 32));
      t = fmaxf(t, __shfl_xor(t, 2, 32));
      t = fmaxf(t, __shfl_xor(t, 4, 32));
      t = fmaxf(t, __shfl_xor(t, 8, 32));
      float mn = fmaxf(mrow[r], t);
      alpha[r] = __expf(mrow[r] - mn);
      mrow[r]  = mn;
      sc[0][r] = __expf(sc[0][r] - mn);
      sc[1][r] = __expf(sc[1][r] - mn);
      float s = sc[0][r] + sc[1][r];
      s += __shfl_xor(s, 1, 32);
      s += __shfl_xor(s, 2, 32);
      s += __shfl_xor(s, 4, 32);
      s += __shfl_xor(s, 8, 32);
      lrow[r] = lrow[r] * alpha[r] + s;
    }
#pragma unroll
    for (int nt = 0; nt < 4; ++nt)
#pragma unroll
      for (int r = 0; r < 8; ++r) oacc[nt][r] *= alpha[r];

    // P (f32 D-layout) -> LDS bf16 -> A-fragment layout (wave-private, DS in-order)
#pragma unroll
    for (int f = 0; f < 2; ++f)
#pragma unroll
      for (int r = 0; r < 8; ++r)
        Pl[pbase + (half * 8 + r) * 32 + f * 16 + l16] = f32_to_bf16(sc[f][r]);
    asm volatile("s_wait_dscnt 0" ::: "memory");
    FragU pu;
#pragma unroll
    for (int j = 0; j < 8; ++j) {
      int kk = half * 16 + 2 * j;
      pu.u[j] = *(const unsigned int*)&Pl[pbase + l16 * 32 + kk];
    }
    v16bf pa = pu.v;

    // O += P * V  (B frag from LDS: col = d, contraction = key, contiguous)
#pragma unroll
    for (int nt = 0; nt < 4; ++nt) {
      int drow = nt * 16 + l16;
      FragU u;
#pragma unroll
      for (int j = 0; j < 8; ++j)
        u.u[j] = *(const unsigned int*)&Vs[cur][drow * 32 + half * 16 + 2 * j];
      oacc[nt] = __builtin_amdgcn_wmma_f32_16x16x32_bf16(false, pa, false, u.v,
                                                         (short)0, oacc[nt], false, false);
    }
    __syncthreads();                            // all waves done with tile kt's LDS
  }

  // normalize and write O in [B,S,E] bf16
  const int b = bh >> 4, h = bh & 15;
#pragma unroll
  for (int r = 0; r < 8; ++r) {
    float inv = 1.0f / lrow[r];
    int qrow  = q0 + half * 8 + r;
#pragma unroll
    for (int nt = 0; nt < 4; ++nt) {
      int col = h * HD + nt * 16 + l16;
      O[((size_t)b * S_LEN + qrow) * 1024 + col] = f32_to_bf16(oacc[nt][r] * inv);
    }
  }
}

// ---------------------------------------------------------------- launch
extern "C" void kernel_launch(void* const* d_in, const int* in_sizes, int n_in,
                              void* d_out, int out_size, void* d_ws, size_t ws_size,
                              hipStream_t stream) {
  const float* x  = (const float*)d_in[0];
  const float* Wq = (const float*)d_in[1];
  const float* bq = (const float*)d_in[2];
  const float* Wk = (const float*)d_in[3];
  const float* bk = (const float*)d_in[4];
  const float* Wv = (const float*)d_in[5];
  const float* bv = (const float*)d_in[6];
  const float* Wo = (const float*)d_in[7];
  const float* bo = (const float*)d_in[8];

  const int M = 4096, E = 1024;   // M = B*S
  unsigned short* p  = (unsigned short*)d_ws;
  unsigned short* xb  = p; p += (size_t)M * E;
  unsigned short* wqt = p; p += (size_t)E * E;   // transposed bf16 weights [N][K]
  unsigned short* wkt = p; p += (size_t)E * E;
  unsigned short* wvt = p; p += (size_t)E * E;
  unsigned short* wot = p; p += (size_t)E * E;
  unsigned short* Qb  = p; p += (size_t)M * E;   // [BH][S][D]
  unsigned short* Kb  = p; p += (size_t)M * E;   // [BH][S][D]
  unsigned short* Vb  = p; p += (size_t)M * E;   // [BH][D][S]
  unsigned short* Ob  = p; p += (size_t)M * E;   // [B][S][E]

  cvt_f32_to_bf16  <<<2048, 256, 0, stream>>>(x,  xb,  M * E);
  cvt_f32_to_bf16_T<<<1024, 256, 0, stream>>>(Wq, wqt, E, E);
  cvt_f32_to_bf16_T<<<1024, 256, 0, stream>>>(Wk, wkt, E, E);
  cvt_f32_to_bf16_T<<<1024, 256, 0, stream>>>(Wv, wvt, E, E);
  cvt_f32_to_bf16_T<<<1024, 256, 0, stream>>>(Wo, wot, E, E);

  dim3 g(M / BM, E / BN);
  gemm_bf16<<<g, 256, 0, stream>>>(xb, wqt, bq, nullptr, Qb, M, E, E, 1);
  gemm_bf16<<<g, 256, 0, stream>>>(xb, wkt, bk, nullptr, Kb, M, E, E, 1);
  gemm_bf16<<<g, 256, 0, stream>>>(xb, wvt, bv, nullptr, Vb, M, E, E, 2);

  attn_kernel<<<dim3(32, S_LEN / 128), 256, 0, stream>>>(Qb, Kb, Vb, Ob);

  gemm_bf16<<<g, 256, 0, stream>>>(Ob, wot, bo, (float*)d_out, nullptr, M, E, E, 0);
}
